// MultiHeadTransformerPosEmb_51659866637136
// MI455X (gfx1250) — compile-verified, matrix operand order checked by reading, not run on hardware
//
#include <hip/hip_runtime.h>
#include <hip/hip_bf16.h>
#include <math.h>
#include <stdint.h>

typedef __attribute__((ext_vector_type(16))) __bf16 v16bf;
typedef __attribute__((ext_vector_type(8)))  float  v8f;
typedef __attribute__((ext_vector_type(4)))  unsigned int u32x4;
typedef __attribute__((ext_vector_type(8)))  int i32x8;
typedef __attribute__((ext_vector_type(4)))  int i32x4;

#define VOCABP   32001
#define HID      256
#define NHEADS   8
#define SEQ      2048
#define BATCH    8
#define NCHUNK   64
#define CHUNK_T  32          // tokens per chunk (2 m-tiles of 16)
#define NTILES   17          // 16 V n-tiles + 1 score n-tile

// workspace layout (float offsets)
#define WS_Y0    0            // 256
#define WS_Q0S   256          // 256
#define WS_R16   512          // 16*256
#define WS_Z     4608         // 8*256
#define WS_OFL   6656         // 8*256
#define WS_PART  8704         // 8*8*64*34 = 139264
#define WS_FRAG  147968       // bf16 frags start here (byte off 591872, 32B aligned)

// ---- fragment index helpers (16-bit WMMA operand layout, wave32) ----
__device__ __forceinline__ int frag_kk(int lane, int j) {
    return (j & 7) | ((j & 8) << 1) | ((lane >> 4) << 3);
}

// ---- TDM: 2D f32 tile (rows x 256) from global -> LDS via Tensor Data Mover ----
__device__ __forceinline__ void tdm_load_tile_f32(const float* gsrc, uint32_t lds_addr,
                                                  int rows_remaining) {
    uint64_t ga = (uint64_t)(uintptr_t)gsrc;
    u32x4 g0;
    g0[0] = 1u;                                        // count=1, user mode
    g0[1] = lds_addr;                                  // LDS byte address
    g0[2] = (uint32_t)ga;                              // global_addr[31:0]
    g0[3] = (uint32_t)((ga >> 32) & 0x01FFFFFFu) | (2u << 30);  // addr[56:32] | type=2

    uint32_t td0 = 256, td1 = (uint32_t)rows_remaining;
    uint32_t tile0 = 256, tile1 = 64;
    i32x8 g1;
    g1[0] = (int)(2u << 16);                           // data_size=4B; no mask/pad/iter
    g1[1] = (int)((td0 & 0xFFFFu) << 16);              // tensor_dim0[15:0] at [63:48]
    g1[2] = (int)(((td0 >> 16) & 0xFFFFu) | ((td1 & 0xFFFFu) << 16));
    g1[3] = (int)(((td1 >> 16) & 0xFFFFu) | (tile0 << 16));   // tile_dim0
    g1[4] = (int)(tile1 & 0xFFFFu);                    // tile_dim1 (tile_dim2=0)
    g1[5] = (int)256;                                  // tensor_dim0_stride[31:0]
    g1[6] = 0;                                         // stride0 hi | stride1 lo
    g1[7] = 0;                                         // tensor_dim1_stride hi

    i32x4 g2; g2[0] = 1; g2[1] = 1; g2[2] = 0; g2[3] = 0;   // tensor_dim2/3=1, tile_dim3=0
    i32x4 g3; g3[0] = 0; g3[1] = (int)(1u << 16); g3[2] = 0; g3[3] = 0; // tensor_dim4=1
#if __clang_major__ >= 23
    i32x8 g4 = {0, 0, 0, 0, 0, 0, 0, 0};
    __builtin_amdgcn_tensor_load_to_lds(g0, g1, g2, g3, g4, 0);
#else
    __builtin_amdgcn_tensor_load_to_lds(g0, g1, g2, g3, 0);
#endif
}

// ================= kernel 1: y0, q0s, r =================
__global__ void prep_kernel(const float* __restrict__ emb,
                            const float* __restrict__ wq,
                            const float* __restrict__ wk,
                            float* __restrict__ ws) {
    __shared__ float y0s[HID], q0ss[HID];
    int h = threadIdx.x;
    float y0 = emb[2 * HID + h] + ((h & 1) ? 1.0f : 0.0f);   // pe(0,h): cos0=1, sin0=0
    y0s[h] = y0; ws[WS_Y0 + h] = y0;
    __syncthreads();
    float acc = 0.f;
    const float* wrow = wq + h * HID;                        // h == n*32+k
    for (int d = 0; d < HID; ++d) acc += y0s[d] * wrow[d];
    float q = acc * 0.17677669529663687f;                    // 1/sqrt(32)
    q0ss[h] = q; ws[WS_Q0S + h] = q;
    __syncthreads();
    for (int n = 0; n < NHEADS; ++n) {
        float r = 0.f;
        for (int k = 0; k < 32; ++k) r += wk[(n * 32 + k) * HID + h] * q0ss[n * 32 + k];
        ws[WS_R16 + n * HID + h] = r;
    }
    for (int n = NHEADS; n < 16; ++n) ws[WS_R16 + n * HID + h] = 0.f;
}

// ================= kernel 2: pack [Wv^T | R^T] into bf16 B-fragments =================
__global__ void conv_kernel(const float* __restrict__ wv,
                            const float* __restrict__ r16,
                            __bf16* __restrict__ frag) {
    int gid = blockIdx.x * 256 + threadIdx.x;   // [nt(17)][ks(8)][lane(32)][j(16)]
    int j = gid & 15, lane = (gid >> 4) & 31, ks = (gid >> 9) & 7, nt = gid >> 12;
    int kk = frag_kk(lane, j);
    int nn = lane & 15;
    int h  = ks * 32 + kk;
    float v = (nt < 16) ? wv[(nt * 16 + nn) * HID + h] : r16[nn * HID + h];
    frag[gid] = (__bf16)v;
}

// ================= kernel 3: fused gather + [V|scores] WMMA + online softmax =================
__global__ void attn_kernel(const int* __restrict__ x,
                            const float* __restrict__ emb,
                            const __bf16* __restrict__ frag,
                            float* __restrict__ ws) {
    __shared__ __bf16 ybf[2 * 8 * 32 * 16];   // A-fragment-swizzled Y tile (16 KB)
    __shared__ float  vc[CHUNK_T * HID];      // V rows (32 KB)
    __shared__ float  sc[CHUNK_T * 16];       // scores (2 KB)
    int chunk = blockIdx.x, b = blockIdx.y;
    int tid = threadIdx.x, lane = tid & 31, wave = tid >> 5;

    // phase 1: gather embedding + positional encoding -> swizzled bf16 LDS
    for (int e = 0; e < 64; ++e) {
        int idx  = tid + e * 128;             // 32*256 elements
        int tloc = idx >> 8, h = idx & 255;
        int p    = chunk * CHUNK_T + tloc;
        int tok  = (p == 0) ? 2 : x[b * (SEQ - 1) + p - 1];
        float freq = __expf((float)(h & ~1) * (-0.0359778921f));  // -ln(10000)/256
        float ang  = (float)p * freq;
        float pe   = (h & 1) ? __cosf(ang) : __sinf(ang);
        float val  = emb[tok * HID + h] + pe;
        int mtile = tloc >> 4, mm = tloc & 15, ks = h >> 5, kk = h & 31;
        int ln = mm + (((kk >> 3) & 1) << 4);
        int jj = (kk & 7) | ((kk >> 4) << 3);
        ybf[(((mtile * 8 + ks) * 32 + ln) << 4) + jj] = (__bf16)val;
    }
    __syncthreads();

    // phase 2: [V | scores] = Ytile(32x256) @ B(256x272) via WMMA
    for (int job = wave; job < 2 * NTILES; job += 4) {
        int mtile = job & 1, nt = job >> 1;   // mtile == wave&1 (loop-invariant per wave)
        v8f acc = {};
        for (int ks = 0; ks < 8; ++ks) {
            v16bf a  = *(const v16bf*)(ybf  + (((mtile * 8 + ks) * 32 + lane) << 4));
            v16bf bm = *(const v16bf*)(frag + (((nt    * 8 + ks) * 32 + lane) << 4));
            acc = __builtin_amdgcn_wmma_f32_16x16x32_bf16(false, a, false, bm,
                                                          (short)0, acc, false, false);
        }
        int rbase = mtile * 16 + ((lane >> 4) << 3);
        int nn = lane & 15;
        if (nt < 16) {
            for (int r = 0; r < 8; ++r) vc[(rbase + r) * HID + nt * 16 + nn] = acc[r];
        } else {
            for (int r = 0; r < 8; ++r) sc[(rbase + r) * 16 + nn] = acc[r];
        }
    }
    __syncthreads();

    // phase 3: per-head online softmax partials over this chunk
    for (int rep = 0; rep < 2; ++rep) {
        int head = (tid >> 5) + 4 * rep;
        int k = lane;
        float m = -1e30f;
        for (int t = 0; t < CHUNK_T; ++t) m = fmaxf(m, sc[t * 16 + head]);
        float l = 0.f, a = 0.f;
        for (int t = 0; t < CHUNK_T; ++t) {
            float p = __expf(sc[t * 16 + head] - m);
            l += p;
            a += p * vc[t * HID + head * 32 + k];
        }
        float* part = ws + WS_PART + (((b * NHEADS + head) * NCHUNK + chunk) * 34);
        if (k == 0) { part[0] = m; part[1] = l; }
        part[2 + k] = a;
    }
}

// ================= kernel 4: merge chunk partials (LSE combine) =================
__global__ void merge_kernel(float* __restrict__ ws) {
    int bn = blockIdx.x;        // b*8 + head
    int k  = threadIdx.x;       // 0..31
    const float* part = ws + WS_PART + bn * NCHUNK * 34;
    float m = -1e30f;
    for (int c = 0; c < NCHUNK; ++c) m = fmaxf(m, part[c * 34]);
    float L = 0.f, s = 0.f;
    for (int c = 0; c < NCHUNK; ++c) {
        float w = __expf(part[c * 34] - m);
        L += part[c * 34 + 1] * w;
        s += w * part[c * 34 + 2 + k];
    }
    ws[WS_OFL + bn * 32 + k] = s / L;
}

// ================= kernel 5: z = o @ Wo^T + 2*y0 =================
__global__ void proj_kernel(const float* __restrict__ wo, float* __restrict__ ws) {
    __shared__ float o[HID], y0[HID];
    int b = blockIdx.x, h = threadIdx.x;
    o[h]  = ws[WS_OFL + b * HID + h];
    y0[h] = ws[WS_Y0 + h];
    __syncthreads();
    float z = 2.f * y0[h];
    const float* wrow = wo + h * HID;
    for (int d = 0; d < HID; ++d) z += o[d] * wrow[d];
    ws[WS_Z + b * HID + h] = z;
}

// ================= kernel 6: out = z @ Wu^T (TDM-staged WMMA) =================
__global__ void final_kernel(const float* __restrict__ wu,
                             const float* __restrict__ ws,
                             float* __restrict__ out) {
    extern __shared__ __align__(16) char fsm[];
    float*  rawt = (float*)fsm;                 // 64 KB: raw f32 wu tile (TDM dest)
    __bf16* wlds = (__bf16*)(fsm + 65536);      // 32 KB: bf16 B-fragments

    int tid = threadIdx.x, lane = tid & 31, nt = tid >> 5;
    int vbase = blockIdx.x * 64;

    // wave 0 kicks the Tensor Data Mover: 64x256 f32 tile, OOB rows auto-zeroed
    if (tid < 32) {
        tdm_load_tile_f32(wu + (size_t)vbase * HID,
                          (uint32_t)(uintptr_t)rawt, VOCABP - vbase);
    }

    // overlap: build A-fragments of z (rows = batches, padded to 16)
    v16bf afr[8];
    int mrow = lane & 15;
    for (int ks = 0; ks < 8; ++ks) {
        v16bf a;
        for (int j = 0; j < 16; ++j) {
            int kk = frag_kk(lane, j);
            float v = (mrow < 8) ? ws[WS_Z + mrow * HID + ks * 32 + kk] : 0.f;
            a[j] = (__bf16)v;
        }
        afr[ks] = a;
    }

    if (tid < 32) __builtin_amdgcn_s_wait_tensorcnt(0);
    __syncthreads();

    // convert raw f32 tile -> bf16 B-fragment layout (LDS -> LDS)
    for (int e = 0; e < 128; ++e) {
        int idx = tid + e * 128;              // 64*256 elements
        int vp = idx >> 8, h = idx & 255;
        float f = rawt[idx];
        int t4 = vp >> 4, nn = vp & 15, ks = h >> 5, kk = h & 31;
        int ln = nn + (((kk >> 3) & 1) << 4);
        int jj = (kk & 7) | ((kk >> 4) << 3);
        wlds[(((t4 * 8 + ks) * 32 + ln) << 4) + jj] = (__bf16)f;
    }
    __syncthreads();

    v8f acc = {};
    for (int ks = 0; ks < 8; ++ks) {
        v16bf bm = *(const v16bf*)(wlds + (((nt * 8 + ks) * 32 + lane) << 4));
        acc = __builtin_amdgcn_wmma_f32_16x16x32_bf16(false, afr[ks], false, bm,
                                                      (short)0, acc, false, false);
    }
    int nn = lane & 15, mbase = (lane >> 4) << 3;
    for (int r = 0; r < 8; ++r) {
        int m = mbase + r;
        int v = vbase + nt * 16 + nn;
        if (m < BATCH && v < VOCABP) out[m * VOCABP + v] = acc[r];
    }
}

extern "C" void kernel_launch(void* const* d_in, const int* in_sizes, int n_in,
                              void* d_out, int out_size, void* d_ws, size_t ws_size,
                              hipStream_t stream) {
    const int*   x   = (const int*)d_in[0];
    const float* emb = (const float*)d_in[1];
    const float* wq  = (const float*)d_in[2];
    const float* wk  = (const float*)d_in[3];
    const float* wv  = (const float*)d_in[4];
    const float* wo  = (const float*)d_in[5];
    const float* wu  = (const float*)d_in[6];
    float* out = (float*)d_out;
    float* ws  = (float*)d_ws;
    __bf16* frag = (__bf16*)(ws + WS_FRAG);

    prep_kernel <<<1, 256, 0, stream>>>(emb, wq, wk, ws);
    conv_kernel <<<(NTILES * 8 * 32 * 16) / 256, 256, 0, stream>>>(wv, ws + WS_R16, frag);
    attn_kernel <<<dim3(NCHUNK, BATCH), 128, 0, stream>>>(x, emb, frag, ws);
    merge_kernel<<<BATCH * NHEADS, 32, 0, stream>>>(ws);
    proj_kernel <<<BATCH, 256, 0, stream>>>(wo, ws);
    final_kernel<<<(VOCABP + 63) / 64, 128, 98304, stream>>>(wu, ws, out);
}
